// GCN_7782480740787
// MI455X (gfx1250) — matrix operand order repaired and checked by reading, not compiled, not run on hardware
//
#include <hip/hip_runtime.h>
#include <hip/hip_bf16.h>

// ---------------------------------------------------------------------------
// GNN forward on MI455X (gfx1250): bf16 WMMA (v_wmma_f32_16x16x32_bf16),
// fused 2-layer MLPs, LDS-resident hidden tiles, b128 LDS fragment loads,
// bf16-materialized y for edge gathers, atomic-f32 scatter-add.
// ---------------------------------------------------------------------------

#define N_NODES 50000
#define N_EDGES 800000
#define N_FEATS 128
#define E_FEATS 64
#define N_OUT   64
#define E_OUT   32
#define HID     128
#define N_STEPS 3

#define STRIDE  136                 // LDS row stride in bf16 (row = 272 B = 17*16)
#define TILE_E  (128 * STRIDE)      // one 128-row LDS tile (ushort elements)
#define SMEM_BYTES (3 * TILE_E * (int)sizeof(unsigned short))

typedef __attribute__((ext_vector_type(16))) __bf16 v16bf;
typedef __attribute__((ext_vector_type(8)))  float  v8f;
typedef __attribute__((ext_vector_type(4)))  unsigned int u32x4;

struct bfpair { u32x4 lo, hi; };    // 32 bytes == v16bf

// round-to-nearest-even f32 -> bf16 (stored as ushort)
__device__ __forceinline__ unsigned short f2bf(float f) {
    unsigned u = __builtin_bit_cast(unsigned, f);
    unsigned r = u + 0x7FFFu + ((u >> 16) & 1u);
    return (unsigned short)(r >> 16);
}

// --- A-matrix fragment (16x32 bf16, MxK), ISA 7.12.2 layout ---------------
__device__ __forceinline__ v16bf load_a_frag(const unsigned short* t, int m0, int k0) {
    int lane = threadIdx.x & 31;
    int grp  = lane >> 4;
    const unsigned short* rowp = t + (m0 + (lane & 15)) * STRIDE + (grp << 3);
    bfpair p;
    p.lo = *reinterpret_cast<const u32x4*>(rowp + k0);
    p.hi = *reinterpret_cast<const u32x4*>(rowp + k0 + 16);
    return __builtin_bit_cast(v16bf, p);
}

// --- B-matrix fragment (32x16 bf16, KxN) from transposed weights ----------
__device__ __forceinline__ v16bf load_b_frag(const unsigned short* t, int n0, int k0) {
    int lane = threadIdx.x & 31;
    int grp  = lane >> 4;
    const unsigned short* rowp = t + (n0 + (lane & 15)) * STRIDE + (grp << 3);
    bfpair p;
    p.lo = *reinterpret_cast<const u32x4*>(rowp + k0);
    p.hi = *reinterpret_cast<const u32x4*>(rowp + k0 + 16);
    return __builtin_bit_cast(v16bf, p);
}

#define ZERO_V8F {0.f,0.f,0.f,0.f,0.f,0.f,0.f,0.f}

// hidden = relu(in @ W1 + b1) -> hid_t (LDS).  KT = DIN/32.
template <int KT>
__device__ __forceinline__ void mlp_phase1(const unsigned short* __restrict__ in_t,
                                           const unsigned short* __restrict__ w_t,
                                           const float* __restrict__ b1,
                                           unsigned short* __restrict__ hid_t) {
    int tid = threadIdx.x, lane = tid & 31, wave = tid >> 5;
    int grp = lane >> 4, nl = lane & 15, m0 = wave * 16;
    v16bf afr[KT];
#pragma unroll
    for (int t = 0; t < KT; ++t) afr[t] = load_a_frag(in_t, m0, 32 * t);
    for (int nt = 0; nt < 8; ++nt) {
        v16bf bfr[KT];
#pragma unroll
        for (int t = 0; t < KT; ++t) bfr[t] = load_b_frag(w_t, nt * 16, 32 * t);
        v8f acc = ZERO_V8F;
#pragma unroll
        for (int t = 0; t < KT; ++t)
            acc = __builtin_amdgcn_wmma_f32_16x16x32_bf16(false, afr[t], false, bfr[t],
                                                          (short)0, acc, false, false);
        int n = nt * 16 + nl;
        float bias = b1[n];
#pragma unroll
        for (int r = 0; r < 8; ++r) {
            float h = acc[r] + bias;
            h = h > 0.f ? h : 0.f;
            hid_t[(m0 + grp * 8 + r) * STRIDE + n] = f2bf(h);
        }
    }
}

// out = hidden @ W2 + b2 (+resid) -> global.  NT = DOUT/16, K = 128.
template <int NT>
__device__ __forceinline__ void mlp_phase2_store(const unsigned short* __restrict__ hid_t,
                                                 const unsigned short* __restrict__ w_t,
                                                 const float* __restrict__ b2,
                                                 float* __restrict__ OUT,
                                                 int base, int M,
                                                 const float* __restrict__ resid) {
    const int dout = NT * 16;
    int tid = threadIdx.x, lane = tid & 31, wave = tid >> 5;
    int grp = lane >> 4, nl = lane & 15, m0 = wave * 16;
    v16bf afr[4];
#pragma unroll
    for (int t = 0; t < 4; ++t) afr[t] = load_a_frag(hid_t, m0, 32 * t);
    for (int nt = 0; nt < NT; ++nt) {
        v16bf bfr[4];
#pragma unroll
        for (int t = 0; t < 4; ++t) bfr[t] = load_b_frag(w_t, nt * 16, 32 * t);
        v8f acc = ZERO_V8F;
#pragma unroll
        for (int t = 0; t < 4; ++t)
            acc = __builtin_amdgcn_wmma_f32_16x16x32_bf16(false, afr[t], false, bfr[t],
                                                          (short)0, acc, false, false);
        int n = nt * 16 + nl;
        float bias = b2[n];
#pragma unroll
        for (int r = 0; r < 8; ++r) {
            int row = base + m0 + grp * 8 + r;
            if (row < M) {
                float v = acc[r] + bias;
                if (resid) v += resid[row * dout + n];
                OUT[row * dout + n] = v;
            }
        }
    }
}

// eh[nt] += hidden @ W2 + b2  (accumulate edge hidden state in registers)
__device__ __forceinline__ void mlp_phase2_acc(const unsigned short* __restrict__ hid_t,
                                               const unsigned short* __restrict__ w_t,
                                               const float* __restrict__ b2,
                                               v8f eh[8]) {
    int tid = threadIdx.x, lane = tid & 31, wave = tid >> 5;
    int nl = lane & 15, m0 = wave * 16;
    v16bf afr[4];
#pragma unroll
    for (int t = 0; t < 4; ++t) afr[t] = load_a_frag(hid_t, m0, 32 * t);
    for (int nt = 0; nt < 8; ++nt) {
        v16bf bfr[4];
#pragma unroll
        for (int t = 0; t < 4; ++t) bfr[t] = load_b_frag(w_t, nt * 16, 32 * t);
        v8f acc = eh[nt];
#pragma unroll
        for (int t = 0; t < 4; ++t)
            acc = __builtin_amdgcn_wmma_f32_16x16x32_bf16(false, afr[t], false, bfr[t],
                                                          (short)0, acc, false, false);
        float bias = b2[nt * 16 + nl];
#pragma unroll
        for (int r = 0; r < 8; ++r) eh[nt][r] = acc[r] + bias;
    }
}

// ---------------------------------------------------------------------------
// Node-side fused 2-layer MLP: OUT = relu(X@W1+b1)@W2+b2 (+resid). DIN = 128.
// ---------------------------------------------------------------------------
template <int DOUTSHIFT>
__global__ void __launch_bounds__(256)
mlp2_node_kernel(const float* __restrict__ X,
                 const float* __restrict__ W1, const float* __restrict__ b1,
                 const float* __restrict__ W2, const float* __restrict__ b2,
                 const float* __restrict__ resid, float* __restrict__ OUT, int M) {
    constexpr int DOUT = 1 << DOUTSHIFT;
    extern __shared__ unsigned short smem[];
    unsigned short* in_t  = smem;
    unsigned short* w_t   = smem + TILE_E;
    unsigned short* hid_t = smem + 2 * TILE_E;
    int tid = threadIdx.x;
    int base = blockIdx.x * 128;

    for (int i = tid; i < 128 * 128; i += 256) {           // input tile (zero-pad)
        int r = i >> 7, c = i & 127, row = base + r;
        float v = (row < M) ? X[row * 128 + c] : 0.f;
        in_t[r * STRIDE + c] = f2bf(v);
    }
    for (int i = tid; i < 128 * 128; i += 256) {           // W1^T
        int n = i & 127, k = i >> 7;
        w_t[n * STRIDE + k] = f2bf(W1[k * 128 + n]);
    }
    __syncthreads();
    mlp_phase1<4>(in_t, w_t, b1, hid_t);
    __syncthreads();
    for (int i = tid; i < 128 * DOUT; i += 256) {          // W2^T
        int n = i & (DOUT - 1), k = i >> DOUTSHIFT;
        w_t[n * STRIDE + k] = f2bf(W2[k * DOUT + n]);
    }
    __syncthreads();
    mlp_phase2_store<(DOUT >> 4)>(hid_t, w_t, b2, OUT, base, M, resid);
}

// ---------------------------------------------------------------------------
// segment_sum: agg[dst[e]] += y[src[e]]  (b128 load + 4 hardware f32 atomics)
// ---------------------------------------------------------------------------
__global__ void __launch_bounds__(256)
scatter_add_kernel(const float* __restrict__ y, const int* __restrict__ src,
                   const int* __restrict__ dst, float* __restrict__ agg, int nE) {
    int i = blockIdx.x * 256 + threadIdx.x;
    int e = i >> 5, c = (i & 31) << 2;
    if (e < nE) {
        const float4 v = *reinterpret_cast<const float4*>(&y[src[e] * HID + c]);
        float* d = &agg[dst[e] * HID + c];
        __hip_atomic_fetch_add(d + 0, v.x, __ATOMIC_RELAXED, __HIP_MEMORY_SCOPE_AGENT);
        __hip_atomic_fetch_add(d + 1, v.y, __ATOMIC_RELAXED, __HIP_MEMORY_SCOPE_AGENT);
        __hip_atomic_fetch_add(d + 2, v.z, __ATOMIC_RELAXED, __HIP_MEMORY_SCOPE_AGENT);
        __hip_atomic_fetch_add(d + 3, v.w, __ATOMIC_RELAXED, __HIP_MEMORY_SCOPE_AGENT);
    }
}

// ---------------------------------------------------------------------------
// f32 -> bf16 packed convert (4 elements / thread)
// ---------------------------------------------------------------------------
__global__ void __launch_bounds__(256)
f32_to_bf16_kernel(const float* __restrict__ in, unsigned short* __restrict__ out, int n) {
    int j = (blockIdx.x * 256 + threadIdx.x) * 4;
    if (j < n) {
        float4 v = *reinterpret_cast<const float4*>(&in[j]);
        uint2 p;
        p.x = (unsigned)f2bf(v.x) | ((unsigned)f2bf(v.y) << 16);
        p.y = (unsigned)f2bf(v.z) | ((unsigned)f2bf(v.w) << 16);
        *reinterpret_cast<uint2*>(&out[j]) = p;
    }
}

// ---------------------------------------------------------------------------
// Edge kernel staging + compute helpers
// ---------------------------------------------------------------------------
// Stage 128 gathered bf16 rows (128 wide) into in_t: pure 16B copies.
__device__ __forceinline__ void stage_bf16_gather(unsigned short* __restrict__ in_t,
                                                  const unsigned short* __restrict__ Ybf,
                                                  const int* __restrict__ gidx,
                                                  int base, int nE) {
    int tid = threadIdx.x;
    for (int i = tid; i < 128 * 16; i += 256) {            // 128 rows x 16 chunks
        int r = i >> 4, q = i & 15, row = base + r;
        u32x4 v = {0u, 0u, 0u, 0u};
        if (row < nE) {
            int g = gidx[row];
            v = *reinterpret_cast<const u32x4*>(&Ybf[g * HID + q * 8]);
        }
        *reinterpret_cast<u32x4*>(&in_t[r * STRIDE + q * 8]) = v;
    }
}

// Stage 128 f32 rows (DIN wide, contiguous) into in_t with conversion.
template <int DSHIFT>
__device__ __forceinline__ void stage_f32_rows(unsigned short* __restrict__ in_t,
                                               const float* __restrict__ IN,
                                               int base, int nE) {
    constexpr int DIN = 1 << DSHIFT;
    int tid = threadIdx.x;
    for (int i = tid; i < (128 << DSHIFT); i += 256) {
        int r = i >> DSHIFT, c = i & (DIN - 1), row = base + r;
        float v = (row < nE) ? IN[row * DIN + c] : 0.f;
        in_t[r * STRIDE + c] = f2bf(v);
    }
}

// eh += 2-layer MLP of the tile already staged in in_t.
template <int DSHIFT>
__device__ __forceinline__ void edge_accum_from_tile(const float* __restrict__ W1,
                                                     const float* __restrict__ b1,
                                                     const float* __restrict__ W2,
                                                     const float* __restrict__ b2,
                                                     unsigned short* in_t,
                                                     unsigned short* w_t,
                                                     unsigned short* hid_t,
                                                     v8f eh[8]) {
    constexpr int DIN = 1 << DSHIFT;
    int tid = threadIdx.x;
    for (int i = tid; i < (128 << DSHIFT); i += 256) {     // W1^T (DIN x 128)
        int n = i & 127, k = i >> 7;
        w_t[n * STRIDE + k] = f2bf(W1[k * 128 + n]);
    }
    __syncthreads();
    mlp_phase1<(DIN >> 5)>(in_t, w_t, b1, hid_t);
    __syncthreads();
    for (int i = tid; i < 128 * 128; i += 256) {           // W2^T (128 x 128)
        int n = i & 127, k = i >> 7;
        w_t[n * STRIDE + k] = f2bf(W2[k * 128 + n]);
    }
    __syncthreads();
    mlp_phase2_acc(hid_t, w_t, b2, eh);
    __syncthreads();
}

// ---------------------------------------------------------------------------
// Fused edge pipeline: out = l(k1(y[src]) + k2(y[dst]) + k(xe))
// ---------------------------------------------------------------------------
__global__ void __launch_bounds__(256)
edge_kernel(const unsigned short* __restrict__ Ybf, const float* __restrict__ XE,
            const int* __restrict__ src, const int* __restrict__ dst,
            const float* k1W1, const float* k1b1, const float* k1W2, const float* k1b2,
            const float* k2W1, const float* k2b1, const float* k2W2, const float* k2b2,
            const float* kW1,  const float* kb1,  const float* kW2,  const float* kb2,
            const float* lW1,  const float* lb1,  const float* lW2,  const float* lb2,
            float* __restrict__ OUT, int nE) {
    extern __shared__ unsigned short smem[];
    unsigned short* in_t  = smem;
    unsigned short* w_t   = smem + TILE_E;
    unsigned short* hid_t = smem + 2 * TILE_E;
    int tid = threadIdx.x, lane = tid & 31, wave = tid >> 5;
    int grp = lane >> 4, nl = lane & 15, m0 = wave * 16;
    int base = blockIdx.x * 128;

    v8f eh[8];
#pragma unroll
    for (int nt = 0; nt < 8; ++nt) { v8f z = ZERO_V8F; eh[nt] = z; }

    stage_bf16_gather(in_t, Ybf, src, base, nE);
    edge_accum_from_tile<7>(k1W1, k1b1, k1W2, k1b2, in_t, w_t, hid_t, eh);

    stage_bf16_gather(in_t, Ybf, dst, base, nE);
    edge_accum_from_tile<7>(k2W1, k2b1, k2W2, k2b2, in_t, w_t, hid_t, eh);

    stage_f32_rows<6>(in_t, XE, base, nE);
    edge_accum_from_tile<6>(kW1, kb1, kW2, kb2, in_t, w_t, hid_t, eh);

    // spill e_hid (f32 regs) -> LDS bf16; each wave writes its own 16 rows
#pragma unroll
    for (int nt = 0; nt < 8; ++nt)
#pragma unroll
        for (int r = 0; r < 8; ++r)
            in_t[(m0 + grp * 8 + r) * STRIDE + nt * 16 + nl] = f2bf(eh[nt][r]);
    __syncthreads();

    // l MLP: 128 -> 128 (relu) -> 32
    for (int i = tid; i < 128 * 128; i += 256) {
        int n = i & 127, k = i >> 7;
        w_t[n * STRIDE + k] = f2bf(lW1[k * 128 + n]);
    }
    __syncthreads();
    mlp_phase1<4>(in_t, w_t, lb1, hid_t);
    __syncthreads();
    for (int i = tid; i < 128 * 32; i += 256) {
        int n = i & 31, k = i >> 5;
        w_t[n * STRIDE + k] = f2bf(lW2[k * 32 + n]);
    }
    __syncthreads();
    mlp_phase2_store<2>(hid_t, w_t, lb2, OUT, base, nE, nullptr);
}

// ---------------------------------------------------------------------------
extern "C" void kernel_launch(void* const* d_in, const int* in_sizes, int n_in,
                              void* d_out, int out_size, void* d_ws, size_t ws_size,
                              hipStream_t stream) {
    (void)in_sizes; (void)n_in; (void)out_size; (void)ws_size;

    const float* x  = (const float*)d_in[0];
    const float* xe = (const float*)d_in[1];
    const int*   ei = (const int*)d_in[2];
    const int* src = ei;
    const int* dst = ei + N_EDGES;

    // MLP params: f g h k1 k2 k l, each (W1,b1,W2,b2) starting at d_in[3]
    const float* P[7][4];
    for (int m = 0; m < 7; ++m)
        for (int j = 0; j < 4; ++j)
            P[m][j] = (const float*)d_in[3 + 4 * m + j];
    const float **fP = P[0], **gP = P[1], **hP = P[2],
                **k1P = P[3], **k2P = P[4], **kP = P[5], **lP = P[6];

    float* yA  = (float*)d_ws;
    float* yB  = yA + (size_t)N_NODES * HID;
    float* agg = yB + (size_t)N_NODES * HID;
    unsigned short* yBf = (unsigned short*)(agg + (size_t)N_NODES * HID);

    float* node_out = (float*)d_out;
    float* edge_out = node_out + (size_t)N_NODES * N_OUT;

    const int nodeBlocks = (N_NODES + 127) / 128;   // 391
    const int edgeBlocks = N_EDGES / 128;           // 6250

    // 1. y = f(x)
    mlp2_node_kernel<7><<<nodeBlocks, 256, SMEM_BYTES, stream>>>(
        x, fP[0], fP[1], fP[2], fP[3], nullptr, yA, N_NODES);

    // 2. message passing: y <- g(segment_sum(y[src], dst)) + y
    float* cur = yA; float* nxt = yB;
    for (int s = 0; s < N_STEPS; ++s) {
        hipMemsetAsync(agg, 0, (size_t)N_NODES * HID * sizeof(float), stream);
        scatter_add_kernel<<<(N_EDGES * 32) / 256, 256, 0, stream>>>(
            cur, src, dst, agg, N_EDGES);
        mlp2_node_kernel<7><<<nodeBlocks, 256, SMEM_BYTES, stream>>>(
            agg, gP[0], gP[1], gP[2], gP[3], cur, nxt, N_NODES);
        float* t = cur; cur = nxt; nxt = t;
    }

    // 3. bf16 copy of final y for edge-side gathers (halves gather traffic)
    f32_to_bf16_kernel<<<(N_NODES * HID) / (256 * 4), 256, 0, stream>>>(
        cur, yBf, N_NODES * HID);

    // 4. node_out = h(y)
    mlp2_node_kernel<6><<<nodeBlocks, 256, SMEM_BYTES, stream>>>(
        cur, hP[0], hP[1], hP[2], hP[3], nullptr, node_out, N_NODES);

    // 5. edge_out = l(k1(y[src]) + k2(y[dst]) + k(xe))
    edge_kernel<<<edgeBlocks, 256, SMEM_BYTES, stream>>>(
        yBf, xe, src, dst,
        k1P[0], k1P[1], k1P[2], k1P[3],
        k2P[0], k2P[1], k2P[2], k2P[3],
        kP[0],  kP[1],  kP[2],  kP[3],
        lP[0],  lP[1],  lP[2],  lP[3],
        edge_out, N_EDGES);
}
